// MambaMixer_59047210385661
// MI455X (gfx1250) — matrix operand order repaired and checked by reading, not compiled, not run on hardware
//
#include <hip/hip_runtime.h>

// ---------------- problem constants (from reference) ----------------
constexpr int Bc = 2, Lc = 1024, Hc = 2048, Ec = 4096, Nc = 16, TRc = 128, Kc = 4;
constexpr int Tc = Bc * Lc;          // 2048 tokens
constexpr int E2 = 2 * Ec;           // 8192
constexpr int SSMW = 160;            // TR + 2N (logical)
constexpr int SSMWP = 192;           // padded to a multiple of 64 for the GEMM

#define USE_ASYNC_LDS 1

// ---------------- vector types ----------------
typedef __attribute__((ext_vector_type(16))) __bf16 v16bf;
typedef __attribute__((ext_vector_type(8)))  __bf16 v8bf;
typedef __attribute__((ext_vector_type(8)))  float  v8f;
typedef __attribute__((ext_vector_type(8)))  unsigned short ushort8v;

static __device__ __forceinline__ unsigned short f32_to_bf16_bits(float f) {
    unsigned u = __builtin_bit_cast(unsigned, f);
    u += 0x7FFFu + ((u >> 16) & 1u);              // round-to-nearest-even
    return (unsigned short)(u >> 16);
}

// ---------------- generic f32 -> bf16 convert ----------------
__global__ void cvt_f32_bf16(const float* __restrict__ x,
                             unsigned short* __restrict__ y, int n) {
    int i = blockIdx.x * 256 + threadIdx.x;
    if (i < n) y[i] = f32_to_bf16_bits(x[i]);
}

// ---------------- W_x convert with zero padding 160 -> 192 cols ----------------
__global__ void cvt_wx_pad(const float* __restrict__ wx,
                           unsigned short* __restrict__ y) {
    int i = blockIdx.x * 256 + threadIdx.x;           // over E * 192
    if (i >= Ec * SSMWP) return;
    int e = i / SSMWP, j = i - e * SSMWP;
    y[i] = (j < SSMW) ? f32_to_bf16_bits(wx[(size_t)e * SSMW + j]) : (unsigned short)0;
}

// ---------------- tiled bf16 WMMA GEMM: C(MxN) = A(MxK) * B(KxN) ----------------
// A,B bf16 row-major; C f32 row-major.
// REQUIRES: M % 128 == 0, N % 64 == 0, K % 32 == 0 (guaranteed by padding).
#define TM 128
#define TN 64
#define TKK 32
#define ASTR 40   // LDS row stride (elements), multiple of 8 -> 16B-aligned chunks
#define BSTR 40

__global__ __launch_bounds__(256)
void gemm_bf16_wmma(const unsigned short* __restrict__ A,
                    const unsigned short* __restrict__ B,
                    float* __restrict__ C, int M, int N, int K) {
    __shared__ __align__(16) unsigned short ldsA[TM * ASTR];
    __shared__ __align__(16) unsigned short ldsB[TN * BSTR];

    const int tid  = threadIdx.x;
    const int lane = tid & 31;
    const int wave = tid >> 5;      // 0..7
    const int wm   = wave >> 1;     // 0..3  (M sub-tile)
    const int wn   = wave & 1;      // 0..1  (N sub-tile)
    const int M0   = blockIdx.y * TM;
    const int N0   = blockIdx.x * TN;

    v8f acc[2][2] = {};

    // cooperative-load mapping
    const int arow = tid >> 1;            // 0..127
    const int acol = (tid & 1) * 16;      // 0 or 16
    const int bk   = tid >> 3;            // 0..31
    const int bn0  = (tid & 7) * 8;       // 0..56

    const unsigned short* aSrc = A + (size_t)(M0 + arow) * K + acol;
    const unsigned short* bSrc = B + (size_t)bk * N + N0 + bn0;
    const size_t bStep = (size_t)TKK * N;     // elements per K-step
#if USE_ASYNC_LDS
    const unsigned ldsAoff0 =
        (unsigned)(unsigned long long)(const void*)&ldsA[arow * ASTR + acol];
#endif

    for (int k0 = 0; k0 < K; k0 += TKK) {
        // ---- stage A tile (row-major) ----
#if USE_ASYNC_LDS
        {
            // CDNA5 async copy: memory -> LDS without VGPR round-trip (ASYNCcnt)
            asm volatile("global_load_async_to_lds_b128 %0, %1, off"
                         :: "v"(ldsAoff0), "v"((const void*)aSrc) : "memory");
            asm volatile("global_load_async_to_lds_b128 %0, %1, off"
                         :: "v"(ldsAoff0 + 16u), "v"((const void*)(aSrc + 8)) : "memory");
        }
#else
        {
            ushort8v v0 = *(const ushort8v*)(aSrc);
            ushort8v v1 = *(const ushort8v*)(aSrc + 8);
            *(ushort8v*)&ldsA[arow * ASTR + acol]     = v0;
            *(ushort8v*)&ldsA[arow * ASTR + acol + 8] = v1;
        }
#endif
        // ---- stage B tile transposed: ldsB[n][k] ----
        {
            ushort8v bv = *(const ushort8v*)bSrc;
            #pragma unroll
            for (int i = 0; i < 8; ++i)
                ldsB[(bn0 + i) * BSTR + bk] = bv[i];
        }
        // prefetch next K tile while we compute (global_prefetch_b8)
        if (k0 + TKK < K) {
            __builtin_prefetch(aSrc + TKK, 0, 1);
            __builtin_prefetch(bSrc + bStep, 0, 1);
        }
        aSrc += TKK;
        bSrc += bStep;
#if USE_ASYNC_LDS
        asm volatile("s_wait_asynccnt 0x0" ::: "memory");
#endif
        __syncthreads();

        const int m  = lane & 15;
        const int ka = (lane < 16) ? 0 : 8;    // A: lanes 0-15 K{0-7,16-23}, 16-31 K{8-15,24-31}
        const int kb = (lane < 16) ? 0 : 8;    // B: lanes 0-15 K{0-7,16-23}, 16-31 K{8-15,24-31}
        #pragma unroll
        for (int ti = 0; ti < 2; ++ti) {
            const int row = wm * 32 + ti * 16 + m;
            v8bf alo = *(const v8bf*)&ldsA[row * ASTR + ka];
            v8bf ahi = *(const v8bf*)&ldsA[row * ASTR + ka + 16];
            v16bf afrag;
            #pragma unroll
            for (int i = 0; i < 8; ++i) { afrag[i] = alo[i]; afrag[i + 8] = ahi[i]; }
            #pragma unroll
            for (int tj = 0; tj < 2; ++tj) {
                const int col = wn * 32 + tj * 16 + m;
                v8bf blo = *(const v8bf*)&ldsB[col * BSTR + kb];
                v8bf bhi = *(const v8bf*)&ldsB[col * BSTR + kb + 16];
                v16bf bfrag;
                #pragma unroll
                for (int i = 0; i < 8; ++i) { bfrag[i] = blo[i]; bfrag[i + 8] = bhi[i]; }
                acc[ti][tj] = __builtin_amdgcn_wmma_f32_16x16x32_bf16(
                    false, afrag, false, bfrag, (short)0, acc[ti][tj], false, false);
            }
        }
        __syncthreads();
    }

    // ---- store C (D layout: VGPR j -> row j (lanes 0-15) / row 8+j (lanes 16-31)) ----
    const int n  = lane & 15;
    const int mo = (lane < 16) ? 0 : 8;
    #pragma unroll
    for (int ti = 0; ti < 2; ++ti)
        #pragma unroll
        for (int tj = 0; tj < 2; ++tj) {
            const int col = N0 + wn * 32 + tj * 16 + n;
            #pragma unroll
            for (int j = 0; j < 8; ++j) {
                const int row = M0 + wm * 32 + ti * 16 + mo + j;
                C[(size_t)row * N + col] = acc[ti][tj][j];
            }
        }
}

// ---------------- depthwise causal conv(K=4) + bias + SiLU ----------------
// proj: [T][2E] f32 (token-major). hidden channel e = col e.
__global__ __launch_bounds__(256)
void conv_silu_kernel(const float* __restrict__ proj, const float* __restrict__ ck,
                      const float* __restrict__ cb, float* __restrict__ hid,
                      unsigned short* __restrict__ hidb) {
    const long idx = (long)blockIdx.x * 256 + threadIdx.x;   // over B*L*E
    if (idx >= (long)Bc * Lc * Ec) return;
    const int e = (int)(idx & (Ec - 1));
    const long tl = idx >> 12;            // log2(E)=12
    const int l = (int)(tl & (Lc - 1));
    const int b = (int)(tl >> 10);        // log2(L)=10
    float acc = cb[e];
    #pragma unroll
    for (int k = 0; k < Kc; ++k) {
        const int ls = l - (Kc - 1) + k;
        if (ls >= 0) acc += proj[(size_t)(b * Lc + ls) * E2 + e] * ck[e * Kc + k];
    }
    const float s = acc / (1.f + __expf(-acc));   // SiLU
    hid[idx]  = s;
    hidb[idx] = f32_to_bf16_bits(s);
}

// ---------------- slice dt (first 128 cols of padded ssm) -> bf16 ----------------
__global__ void dt_slice_bf16(const float* __restrict__ ssm,
                              unsigned short* __restrict__ dtb) {
    const int idx = blockIdx.x * 256 + threadIdx.x;   // over T*TR
    if (idx >= Tc * TRc) return;
    const int t = idx >> 7, j = idx & (TRc - 1);
    dtb[idx] = f32_to_bf16_bits(ssm[(size_t)t * SSMWP + j]);
}

// ---------------- softplus(dts + b_dt) in place ----------------
__global__ void softplus_kernel(float* __restrict__ dts, const float* __restrict__ bdt) {
    const long idx = (long)blockIdx.x * 256 + threadIdx.x;  // over T*E
    if (idx >= (long)Tc * Ec) return;
    const int e = (int)(idx & (Ec - 1));
    const float x = dts[idx] + bdt[e];
    dts[idx] = (x > 20.f) ? x : log1pf(__expf(x));
}

// ---------------- selective scan + D skip + SiLU(gate), bf16 out ----------------
__global__ __launch_bounds__(256)
void scan_kernel(const float* __restrict__ dts, const float* __restrict__ hid,
                 const float* __restrict__ ssm, const float* __restrict__ proj,
                 const float* __restrict__ Alog, const float* __restrict__ Dv,
                 unsigned short* __restrict__ yg) {
    __shared__ float bc[16][32];   // 16 timesteps of (Bm[16], Cm[16])
    const int tid = threadIdx.x;
    const int e = blockIdx.x * 256 + tid;
    const int b = blockIdx.y;

    float An[Nc], st[Nc];
    #pragma unroll
    for (int n = 0; n < Nc; ++n) { An[n] = -__expf(Alog[e * Nc + n]); st[n] = 0.f; }
    const float Dval = Dv[e];

    for (int l0 = 0; l0 < Lc; l0 += 16) {
        for (int i = tid; i < 512; i += 256) {
            const int s = i >> 5, j = i & 31;
            bc[s][j] = ssm[(size_t)(b * Lc + l0 + s) * SSMWP + TRc + j];
        }
        __syncthreads();
        #pragma unroll 1
        for (int s = 0; s < 16; ++s) {
            const size_t t = (size_t)b * Lc + l0 + s;
            const float dv = dts[t * Ec + e];
            const float h  = hid[t * Ec + e];
            const float dh = dv * h;
            float y = 0.f;
            #pragma unroll
            for (int n = 0; n < Nc; ++n) {
                const float dA = __expf(dv * An[n]);
                st[n] = dA * st[n] + dh * bc[s][n];
                y += st[n] * bc[s][16 + n];
            }
            const float g  = proj[t * E2 + Ec + e];
            const float sg = g / (1.f + __expf(-g));
            yg[t * Ec + e] = f32_to_bf16_bits((y + h * Dval) * sg);
        }
        __syncthreads();
    }
}

// ---------------- host-side orchestration ----------------
extern "C" void kernel_launch(void* const* d_in, const int* in_sizes, int n_in,
                              void* d_out, int out_size, void* d_ws, size_t ws_size,
                              hipStream_t stream) {
    const float* X     = (const float*)d_in[0];   // (B,L,H)
    const float* Win   = (const float*)d_in[1];   // (H,2E)
    const float* Ck    = (const float*)d_in[2];   // (E,1,K)
    const float* Cb    = (const float*)d_in[3];   // (E)
    const float* Wx    = (const float*)d_in[4];   // (E,160)
    const float* Wdt   = (const float*)d_in[5];   // (TR,E)
    const float* bdt   = (const float*)d_in[6];   // (E)
    const float* Wout  = (const float*)d_in[7];   // (E,H)
    const float* Alog  = (const float*)d_in[8];   // (E,N)
    const float* Dv    = (const float*)d_in[9];   // (E)
    float* out = (float*)d_out;                   // (B,L,H) f32

    char* w = (char*)d_ws;
    size_t off = 0;
    auto alloc = [&](size_t bytes) -> void* {
        void* p = w + off;
        off += (bytes + 255) & ~(size_t)255;
        return p;
    };
    unsigned short* Xb    = (unsigned short*)alloc((size_t)Tc * Hc * 2);
    unsigned short* Winb  = (unsigned short*)alloc((size_t)Hc * E2 * 2);
    float*          proj  = (float*)         alloc((size_t)Tc * E2 * 4);
    float*          hid   = (float*)         alloc((size_t)Tc * Ec * 4);
    unsigned short* hidb  = (unsigned short*)alloc((size_t)Tc * Ec * 2);
    unsigned short* Wxb   = (unsigned short*)alloc((size_t)Ec * SSMWP * 2);
    float*          ssm   = (float*)         alloc((size_t)Tc * SSMWP * 4);
    unsigned short* dtb   = (unsigned short*)alloc((size_t)Tc * TRc * 2);
    unsigned short* Wdtb  = (unsigned short*)alloc((size_t)TRc * Ec * 2);
    float*          dts   = (float*)         alloc((size_t)Tc * Ec * 4);
    unsigned short* ygb   = (unsigned short*)alloc((size_t)Tc * Ec * 2);
    unsigned short* Woutb = (unsigned short*)alloc((size_t)Ec * Hc * 2);
    (void)ws_size; (void)in_sizes; (void)n_in; (void)out_size;

    auto cvt = [&](const float* src, unsigned short* dst, int n) {
        cvt_f32_bf16<<<(n + 255) / 256, 256, 0, stream>>>(src, dst, n);
    };
    auto gemm = [&](const unsigned short* A, const unsigned short* B, float* C,
                    int M, int N, int K) {
        dim3 grid((N + TN - 1) / TN, (M + TM - 1) / TM);
        gemm_bf16_wmma<<<grid, 256, 0, stream>>>(A, B, C, M, N, K);
    };

    // 1. precision converts (W_x is padded 160 -> 192 cols with zeros)
    cvt(X,    Xb,    Tc * Hc);
    cvt(Win,  Winb,  Hc * E2);
    cvt_wx_pad<<<(Ec * SSMWP + 255) / 256, 256, 0, stream>>>(Wx, Wxb);
    cvt(Wdt,  Wdtb,  TRc * Ec);
    cvt(Wout, Woutb, Ec * Hc);

    // 2. in-projection: proj[T][2E] = X @ W_in
    gemm(Xb, Winb, proj, Tc, E2, Hc);

    // 3. depthwise causal conv + SiLU -> hid (f32 + bf16)
    {
        long total = (long)Bc * Lc * Ec;
        conv_silu_kernel<<<(unsigned)((total + 255) / 256), 256, 0, stream>>>(
            proj, Ck, Cb, hid, hidb);
    }

    // 4. ssm params: ssm[T][192] = hid @ W_x_padded
    gemm(hidb, Wxb, ssm, Tc, SSMWP, Ec);

    // 5. dt slice -> bf16; dts[T][E] = dt @ W_dt; softplus(+b_dt)
    dt_slice_bf16<<<(Tc * TRc + 255) / 256, 256, 0, stream>>>(ssm, dtb);
    gemm(dtb, Wdtb, dts, Tc, Ec, TRc);
    {
        long total = (long)Tc * Ec;
        softplus_kernel<<<(unsigned)((total + 255) / 256), 256, 0, stream>>>(dts, bdt);
    }

    // 6. selective scan + gating -> ygb (bf16)
    {
        dim3 grid(Ec / 256, Bc);
        scan_kernel<<<grid, 256, 0, stream>>>(dts, hid, ssm, proj, Alog, Dv, ygb);
    }

    // 7. out-projection straight into d_out: out[T][H] = yg @ W_out
    gemm(ygb, Woutb, out, Tc, Hc, Ec);
}
